// PointNet2feat_6322191859819
// MI455X (gfx1250) — matrix operand order repaired and checked by reading, not compiled, not run on hardware
//
#include <hip/hip_runtime.h>

#define B_   8
#define N_   16384
#define P_   (B_ * N_)        // 131072 points
#define EPSF 1e-5f

typedef __attribute__((ext_vector_type(16))) _Float16 v16h;
typedef __attribute__((ext_vector_type(8)))  _Float16 v8h;
typedef __attribute__((ext_vector_type(8)))  float    v8f;

union V16 { v16h v; v8h h[2]; };
union H2  { _Float16 h[2]; unsigned u; };

// ---------------------------------------------------------------- init
__global__ void pn_init(float* sum0, float* sq0, float* sum1, float* sq1,
                        float* sum2, float* sq2, float* pmax,
                        _Float16* W1h, const float* W1,
                        _Float16* W2h, const float* W2) {
  int i = blockIdx.x * blockDim.x + threadIdx.x;
  if (i < 64)   { sum0[i] = 0.f; sq0[i] = 0.f; }
  if (i < 128)  { sum1[i] = 0.f; sq1[i] = 0.f; }
  if (i < 1024) { sum2[i] = 0.f; sq2[i] = 0.f; }
  if (i < 128 * 64) W1h[i] = (_Float16)W1[i];
  if (i < 1024 * 128) W2h[i] = (_Float16)W2[i];
  if (i < 8 * 16 * 1024) pmax[i] = -__builtin_inff();
}

// ------------------------------------------------- layer0: 3->64 + stats
__global__ void pn_layer0(const float* __restrict__ x, const float* __restrict__ W0,
                          const float* __restrict__ b0, float* __restrict__ h0,
                          float* sum0, float* sq0) {
  __shared__ float lsum[64], lsq[64];
  int t = threadIdx.x;
  if (t < 64) { lsum[t] = 0.f; lsq[t] = 0.f; }
  __syncthreads();
  long start = (long)blockIdx.x * blockDim.x + t;
  long stride = (long)gridDim.x * blockDim.x;      // multiple of 64 -> o fixed
  int o = (int)(start & 63);
  float w0 = W0[o * 3 + 0], w1 = W0[o * 3 + 1], w2 = W0[o * 3 + 2], bb = b0[o];
  float accS = 0.f, accQ = 0.f;
  for (long idx = start; idx < (long)P_ * 64; idx += stride) {
    long p = idx >> 6;
    float v = fmaf(x[p * 3 + 0], w0,
              fmaf(x[p * 3 + 1], w1,
              fmaf(x[p * 3 + 2], w2, bb)));
    h0[idx] = v;
    accS += v; accQ += v * v;
  }
  atomicAdd(&lsum[t & 63], accS);
  atomicAdd(&lsq[t & 63], accQ);
  __syncthreads();
  if (t < 64) { atomicAdd(&sum0[t], lsum[t]); atomicAdd(&sq0[t], lsq[t]); }
}

// ------------------------------------------------- BN finalize (generic)
__global__ void pn_finalize(const float* sum, const float* sq, const float* g,
                            const float* be, float* scale, float* shift,
                            int C, float inv_cnt) {
  int c = blockIdx.x * blockDim.x + threadIdx.x;
  if (c >= C) return;
  float m = sum[c] * inv_cnt;
  float v = sq[c] * inv_cnt - m * m;
  float s = g[c] * rsqrtf(v + EPSF);
  scale[c] = s;
  shift[c] = be[c] - m * s;
}

// --------------------- BN0 + ReLU: write f16 feed + transposed out tail
__global__ void pn_bn0(const float* __restrict__ h0, const float* __restrict__ sc,
                       const float* __restrict__ sh, _Float16* __restrict__ localh,
                       float* __restrict__ out) {
  __shared__ float tile[64 * 65];
  int t = threadIdx.x;
  long p0 = (long)blockIdx.x * 64;
  int b  = (int)(p0 / N_);
  int n0 = (int)(p0 % N_);
  int ch = t & 63;
  float s = sc[ch], q = sh[ch];
  for (int i = 0; i < 16; ++i) {
    int e = t + 256 * i;
    int pt = e >> 6;                       // e&63 == t&63 == ch
    float v = fmaxf(0.f, fmaf(h0[(p0 + pt) * 64 + ch], s, q));
    localh[(p0 + pt) * 64 + ch] = (_Float16)v;
    tile[pt * 65 + ch] = v;
  }
  __syncthreads();
  for (int i = 0; i < 16; ++i) {
    int e = t + 256 * i;
    int c2 = e >> 6, pt = e & 63;
    out[((long)b * 1088 + 1024 + c2) * N_ + n0 + pt] = tile[pt * 65 + c2];
  }
}

// ------------------------------ GEMM1: [P,64]f16 x W1^T -> h1[P,128] + stats
__global__ void pn_gemm1(const _Float16* __restrict__ A, const _Float16* __restrict__ W1h,
                         const float* __restrict__ b1, float* __restrict__ h1,
                         float* sum1, float* sq1) {
  int t = threadIdx.x, w = t >> 5, lane = t & 31;
  int lhalf = lane >> 4, lm = lane & 15;
  int o0[2] = { w * 32, w * 32 + 16 };
  V16 bf[2][2];
  float bias[2], sAcc[2] = {0.f, 0.f}, qAcc[2] = {0.f, 0.f};
  for (int j = 0; j < 2; ++j) {
    int o = o0[j] + lm;
    bias[j] = b1[o];
    for (int ks = 0; ks < 2; ++ks) {
      int kb = ks * 32 + lhalf * 16;              // B: lanes 0-15 K lo, 16-31 K hi
      bf[j][ks].v = *(const v16h*)(W1h + o * 64 + kb);
    }
  }
  for (long tile = blockIdx.x; tile < P_ / 16; tile += gridDim.x) {
    long p0 = tile * 16;
    long arow = (p0 + lm) * 64;
    V16 af[2];
#pragma unroll
    for (int ks = 0; ks < 2; ++ks) {              // A: K = e + 16*lhalf
      int kb = ks * 32 + lhalf * 16;
      af[ks].h[0] = *(const v8h*)(A + arow + kb);
      af[ks].h[1] = *(const v8h*)(A + arow + kb + 8);
    }
#pragma unroll
    for (int j = 0; j < 2; ++j) {
      v8f c = {};
      c = __builtin_amdgcn_wmma_f32_16x16x32_f16(false, af[0].v, false, bf[j][0].v,
                                                 (short)0, c, false, false);
      c = __builtin_amdgcn_wmma_f32_16x16x32_f16(false, af[1].v, false, bf[j][1].v,
                                                 (short)0, c, false, false);
      int o = o0[j] + lm;
#pragma unroll
      for (int i = 0; i < 8; ++i) {               // row p = p0 + i + 8*lhalf
        long p = p0 + i + 8 * lhalf;
        float v = c[i] + bias[j];
        h1[p * 128 + o] = v;
        sAcc[j] += v; qAcc[j] += v * v;
      }
    }
  }
  for (int j = 0; j < 2; ++j) {
    float s = sAcc[j] + __shfl_down(sAcc[j], 16);
    float q = qAcc[j] + __shfl_down(qAcc[j], 16);
    if (lane < 16) {
      atomicAdd(&sum1[o0[j] + lane], s);
      atomicAdd(&sq1[o0[j] + lane], q);
    }
  }
}

// ----------------- BN1 + ReLU elementwise: h1 f32 -> h1n f16 (packed pairs)
__global__ void pn_bn1(const float* __restrict__ h1, const float* __restrict__ sc1,
                       const float* __restrict__ sh1, _Float16* __restrict__ h1n) {
  const long totalPairs = (long)P_ * 64;          // 128 ch / 2 per pair
  long start = (long)blockIdx.x * blockDim.x + threadIdx.x;
  long stride = (long)gridDim.x * blockDim.x;     // multiple of 64 -> cp fixed
  int cp = (int)((start & 63) * 2);
  float s0 = sc1[cp], s1 = sc1[cp + 1], t0 = sh1[cp], t1 = sh1[cp + 1];
  unsigned* dst = (unsigned*)h1n;
  for (long idx = start; idx < totalPairs; idx += stride) {
    float2 hv = *(const float2*)(h1 + idx * 2);
    H2 pk;
    pk.h[0] = (_Float16)fmaxf(0.f, fmaf(hv.x, s0, t0));
    pk.h[1] = (_Float16)fmaxf(0.f, fmaf(hv.y, s1, t1));
    dst[idx] = pk.u;
  }
}

// -------- GEMM2: async-LDS double-buffered staging, WMMA K=128, max+stats
// grid.x = B*16 chunks (1024 pts), grid.y = 4 groups of 256 channels
// block = 256 threads = 8 waves x 2 channel tiles of 16
__global__ void pn_gemm2(const _Float16* __restrict__ h1n,
                         const _Float16* __restrict__ W2h,
                         const float* __restrict__ b2, float* __restrict__ pmax,
                         float* sum2, float* sq2) {
  __shared__ _Float16 ash[2][16 * 136];           // padded rows, conflict-free b128
  int t = threadIdx.x, w = t >> 5, lane = t & 31;
  int lhalf = lane >> 4, lm = lane & 15;
  int bIdx  = blockIdx.x >> 4;
  int chunk = blockIdx.x & 15;
  long pbase = (long)bIdx * N_ + (long)chunk * 1024;

  // staging copy geometry: one b128 (8 halves) per thread covers 16x128 tile
  int spt   = t >> 4;                             // staged row 0..15
  int skoff = (t & 15) * 8;                       // halves within row

  // B fragments: 2 channel tiles x 4 k-steps, preloaded
  V16 bf[2][4];
  float bias[2];
  int oB[2];
#pragma unroll
  for (int j = 0; j < 2; ++j) {
    int o = blockIdx.y * 256 + w * 32 + j * 16 + lm;
    oB[j] = o;
    bias[j] = b2[o];
#pragma unroll
    for (int ks = 0; ks < 4; ++ks)
      bf[j][ks].v = *(const v16h*)(W2h + (long)o * 128 + ks * 32 + lhalf * 16);
  }
  float mAcc[2] = { -__builtin_inff(), -__builtin_inff() };
  float sAcc[2] = { 0.f, 0.f }, qAcc[2] = { 0.f, 0.f };

  // prologue: async-stage tile 0 into buffer 0
  {
    unsigned lds0 = (unsigned)(uintptr_t)&ash[0][spt * 136 + skoff];
    unsigned long long ga =
        (unsigned long long)(uintptr_t)(h1n + (pbase + spt) * 128 + skoff);
    asm volatile("global_load_async_to_lds_b128 %0, %1, off"
                 :: "v"(lds0), "v"(ga) : "memory");
  }
  asm volatile("s_wait_asynccnt 0x0" ::: "memory");
  __syncthreads();

  for (int tile = 0; tile < 64; ++tile) {
    int cur = tile & 1;
    if (tile + 1 < 64) {                          // prefetch next tile
      long p0n = pbase + (long)(tile + 1) * 16;
      unsigned lds1 = (unsigned)(uintptr_t)&ash[cur ^ 1][spt * 136 + skoff];
      unsigned long long ga =
          (unsigned long long)(uintptr_t)(h1n + (p0n + spt) * 128 + skoff);
      asm volatile("global_load_async_to_lds_b128 %0, %1, off"
                   :: "v"(lds1), "v"(ga) : "memory");
    }
    v8f c0 = {}, c1 = {};
    const _Float16* arow = &ash[cur][lm * 136];
#pragma unroll
    for (int ks = 0; ks < 4; ++ks) {              // A shared by both channel tiles
      int kb = ks * 32 + lhalf * 16;
      V16 af;
      af.h[0] = *(const v8h*)(arow + kb);
      af.h[1] = *(const v8h*)(arow + kb + 8);
      c0 = __builtin_amdgcn_wmma_f32_16x16x32_f16(false, af.v, false, bf[0][ks].v,
                                                  (short)0, c0, false, false);
      c1 = __builtin_amdgcn_wmma_f32_16x16x32_f16(false, af.v, false, bf[1][ks].v,
                                                  (short)0, c1, false, false);
    }
#pragma unroll
    for (int i = 0; i < 8; ++i) {
      float v0 = c0[i] + bias[0];
      float v1 = c1[i] + bias[1];
      mAcc[0] = fmaxf(mAcc[0], v0); sAcc[0] += v0; qAcc[0] += v0 * v0;
      mAcc[1] = fmaxf(mAcc[1], v1); sAcc[1] += v1; qAcc[1] += v1 * v1;
    }
    asm volatile("s_wait_asynccnt 0x0" ::: "memory");
    __syncthreads();
  }
#pragma unroll
  for (int j = 0; j < 2; ++j) {
    float mo = fmaxf(mAcc[j], __shfl_down(mAcc[j], 16));
    float so = sAcc[j] + __shfl_down(sAcc[j], 16);
    float qo = qAcc[j] + __shfl_down(qAcc[j], 16);
    if (lane < 16) {
      int oc = blockIdx.y * 256 + w * 32 + j * 16 + lane;
      pmax[((long)bIdx * 16 + chunk) * 1024 + oc] = mo;  // exclusive slot
      atomicAdd(&sum2[oc], so);
      atomicAdd(&sq2[oc], qo);
    }
  }
  (void)oB;
}

// ----------------- finalize layer2 BN + reduce chunk maxes -> gfeat[b][o]
__global__ void pn_gfeat(const float* pmax, const float* sum2, const float* sq2,
                         const float* g2, const float* be2, float* gfeat) {
  int tid = blockIdx.x * blockDim.x + threadIdx.x;
  if (tid >= 8 * 1024) return;
  int b = tid >> 10, o = tid & 1023;
  const float inv = 1.f / (float)P_;
  float m = sum2[o] * inv;
  float v = sq2[o] * inv - m * m;
  float s = g2[o] * rsqrtf(v + EPSF);
  float sh = be2[o] - m * s;
  float mx = -__builtin_inff();
  for (int c = 0; c < 16; ++c)
    mx = fmaxf(mx, pmax[((long)b * 16 + c) * 1024 + o]);
  gfeat[tid] = fmaf(mx, s, sh);                   // s>0: max commutes with BN
}

// ------------------------------------ broadcast gfeat over N (536 MB head)
__global__ void pn_broadcast(const float* __restrict__ gfeat, float* __restrict__ out) {
  const long QN = N_ / 4;
  const long total = (long)B_ * 1024 * QN;
  float4* o4 = (float4*)out;
  for (long idx = (long)blockIdx.x * blockDim.x + threadIdx.x; idx < total;
       idx += (long)gridDim.x * blockDim.x) {
    long bo = idx / QN;                           // b*1024 + o
    long nq = idx - bo * QN;
    int b = (int)(bo >> 10), o = (int)(bo & 1023);
    float v = gfeat[bo];
    float4 f = {v, v, v, v};
    o4[((long)b * 1088 + o) * QN + nq] = f;
  }
}

extern "C" void kernel_launch(void* const* d_in, const int* in_sizes, int n_in,
                              void* d_out, int out_size, void* d_ws, size_t ws_size,
                              hipStream_t stream) {
  const float* x   = (const float*)d_in[0];
  const float* W0  = (const float*)d_in[1];
  const float* b0  = (const float*)d_in[2];
  const float* g0  = (const float*)d_in[3];
  const float* be0 = (const float*)d_in[4];
  const float* W1  = (const float*)d_in[5];
  const float* b1  = (const float*)d_in[6];
  const float* g1  = (const float*)d_in[7];
  const float* be1 = (const float*)d_in[8];
  const float* W2  = (const float*)d_in[9];
  const float* b2  = (const float*)d_in[10];
  const float* g2  = (const float*)d_in[11];
  const float* be2 = (const float*)d_in[12];
  float* out = (float*)d_out;

  char* ws = (char*)d_ws;
  float*    h0     = (float*)ws;      ws += (size_t)P_ * 64 * 4;
  float*    h1     = (float*)ws;      ws += (size_t)P_ * 128 * 4;
  _Float16* localh = (_Float16*)ws;   ws += (size_t)P_ * 64 * 2;
  _Float16* h1n    = (_Float16*)ws;   ws += (size_t)P_ * 128 * 2;
  _Float16* W1h    = (_Float16*)ws;   ws += 128 * 64 * 2;
  _Float16* W2h    = (_Float16*)ws;   ws += 1024 * 128 * 2;
  float* sum0 = (float*)ws; ws += 64 * 4;
  float* sq0  = (float*)ws; ws += 64 * 4;
  float* sc0  = (float*)ws; ws += 64 * 4;
  float* sh0  = (float*)ws; ws += 64 * 4;
  float* sum1 = (float*)ws; ws += 128 * 4;
  float* sq1  = (float*)ws; ws += 128 * 4;
  float* sc1  = (float*)ws; ws += 128 * 4;
  float* sh1  = (float*)ws; ws += 128 * 4;
  float* sum2 = (float*)ws; ws += 1024 * 4;
  float* sq2  = (float*)ws; ws += 1024 * 4;
  float* pmax = (float*)ws; ws += (size_t)8 * 16 * 1024 * 4;
  float* gfeat = (float*)ws; ws += 8 * 1024 * 4;

  const float invP = 1.f / (float)P_;

  pn_init<<<512, 256, 0, stream>>>(sum0, sq0, sum1, sq1, sum2, sq2, pmax,
                                   W1h, W1, W2h, W2);
  pn_layer0<<<2048, 256, 0, stream>>>(x, W0, b0, h0, sum0, sq0);
  pn_finalize<<<1, 64, 0, stream>>>(sum0, sq0, g0, be0, sc0, sh0, 64, invP);
  pn_bn0<<<P_ / 64, 256, 0, stream>>>(h0, sc0, sh0, localh, out);
  pn_gemm1<<<1024, 128, 0, stream>>>(localh, W1h, b1, h1, sum1, sq1);
  pn_finalize<<<1, 128, 0, stream>>>(sum1, sq1, g1, be1, sc1, sh1, 128, invP);
  pn_bn1<<<2048, 256, 0, stream>>>(h1, sc1, sh1, h1n);
  pn_gemm2<<<dim3(128, 4), 256, 0, stream>>>(h1n, W2h, b2, pmax, sum2, sq2);
  pn_gfeat<<<32, 256, 0, stream>>>(pmax, sum2, sq2, g2, be2, gfeat);
  pn_broadcast<<<2048, 256, 0, stream>>>(gfeat, out);
}
